// CCNN2d_55018531061964
// MI455X (gfx1250) — matrix operand order repaired and checked by reading, not compiled, not run on hardware
//
#include <hip/hip_runtime.h>
#include <hip/hip_bf16.h>

// ---------------------------------------------------------------------------
// CCNN2d recurrent conv net for MI455X (gfx1250, wave32, WMMA).
//   f = DF*f + conv(y, W) + b + x_t      (32x32x3x3 learned conv -> WMMA GEMM)
//   l = DL*l + stencil3x3(sum_c y)       (fixed kernel identical per channel
//                                         pair -> collapses to channel-sum)
//   u = f + 0.5*f*l ;  e = DE*e + 10*y ;  y = sigmoid(u - e)
// ---------------------------------------------------------------------------

#define Bb 8
#define Tt 10
#define Cc 32
#define Hh 128
#define Ww 128
#define HWp (Hh * Ww)

#define DF 0.9048374180359595f   /* exp(-0.1) */
#define DL 0.36787944117144233f  /* exp(-1.0) */
#define DE 0.36787944117144233f  /* exp(-1.0) */
#define VE 10.0f

typedef __bf16 bf16;
typedef bf16  v16bf __attribute__((ext_vector_type(16)));
typedef float v8f   __attribute__((ext_vector_type(8)));

// Tile: 8 rows x 32 cols of pixels per block (+1 halo each side).
#define TH 8
#define TW 32
#define LH (TH + 2)
#define LW (TW + 2)

// ---------------------------------------------------------------------------
// Init recurrent state: f = 0, l = 0, e = V_E/ALPHA_E = 10.
// ---------------------------------------------------------------------------
__global__ __launch_bounds__(256) void init_state_kernel(float* __restrict__ f,
                                                         float* __restrict__ l,
                                                         float* __restrict__ e) {
    size_t i = (size_t)blockIdx.x * 256 + threadIdx.x;
    if (i < (size_t)Bb * Cc * HWp) {
        f[i] = 0.0f;
        l[i] = 0.0f;
        e[i] = 10.0f;
    }
}

// ---------------------------------------------------------------------------
// Pre-swizzle conv weights into bf16 WMMA A-fragment layout.
// wA layout: [tap][mhalf][lane][slot j=0..15], one v16bf per lane.
// A-matrix (16x32, MxK) layout used here (contiguous K, not AMD's table order): lane m (0..15 / 16..31 same M),
//   slots 0..15 -> K = 16*(lane>>4) + j
// ---------------------------------------------------------------------------
__global__ __launch_bounds__(256) void prep_weights_kernel(const float* __restrict__ w,
                                                           bf16* __restrict__ wA) {
    int idx = blockIdx.x * 256 + threadIdx.x;
    if (idx >= 9 * 2 * 32 * 16) return;
    int j    = idx & 15;
    int lane = (idx >> 4) & 31;
    int half = (idx >> 9) & 1;
    int tap  = idx >> 10;                       // 0..8 = ky*3 + kx
    int m    = half * 16 + (lane & 15);         // output channel
    int kb   = 16 * (lane >> 4);
    int cin  = kb + j;
    wA[idx] = (bf16)w[((size_t)m * Cc + cin) * 9 + tap];
}

// ---------------------------------------------------------------------------
// Channel sum of previous y: s[b,h,w] = sum_c y[b, t-1, c, h, w]
// ---------------------------------------------------------------------------
__global__ __launch_bounds__(256) void chan_sum_kernel(const float* __restrict__ out,
                                                       float* __restrict__ s, int tm1) {
    size_t i = (size_t)blockIdx.x * 256 + threadIdx.x;
    if (i >= (size_t)Bb * HWp) return;
    int b  = (int)(i / HWp);
    int hw = (int)(i % HWp);
    const float* p = out + ((size_t)(b * Tt + tm1)) * Cc * HWp + hw;
    float a = 0.0f;
#pragma unroll
    for (int c = 0; c < Cc; ++c) a += p[(size_t)c * HWp];
    s[i] = a;
}

// ---------------------------------------------------------------------------
// Fused recurrent step. One block = one 8x32 pixel tile of one batch image.
// 8 wave32's; each wave owns one pixel row and produces all 32 channels for
// two groups of 16 pixels via v_wmma_f32_16x16x32_bf16 (9 taps x 2 M-halves).
// ---------------------------------------------------------------------------
__global__ __launch_bounds__(256) void step_kernel(const float* __restrict__ x,
                                                   const float* __restrict__ bias,
                                                   const bf16* __restrict__ wA,
                                                   const float* __restrict__ s,
                                                   float* __restrict__ out,
                                                   float* __restrict__ fbuf,
                                                   float* __restrict__ lbuf,
                                                   float* __restrict__ ebuf,
                                                   int t, int first) {
    const int b   = blockIdx.z;
    const int h0  = blockIdx.y * TH;
    const int w0  = blockIdx.x * TW;
    const int tid = threadIdx.x;
    const int lane = tid & 31;
    const int wv   = tid >> 5;

    // LDS: y_prev tile + halo, bf16, channel-fastest (c0 multiple of 16 -> 32B
    // aligned v16bf reads -> ds_load_b128 pairs for WMMA B fragments).
    __shared__ alignas(32) bf16 yt[LH * LW * Cc];

    const float* xt  = x + ((size_t)(b * Tt + t)) * Cc * HWp;
    float*       yob = out + ((size_t)(b * Tt + t)) * Cc * HWp;
    const float* ypb = out + ((size_t)(b * Tt + (t > 0 ? t - 1 : 0))) * Cc * HWp;

    if (!first) {
        // Stage y_prev (f32 -> bf16) into LDS, w-fastest for coalescing.
        for (int idx = tid; idx < LH * LW * Cc; idx += 256) {
            int ww  = idx % LW;
            int tmp = idx / LW;
            int c   = tmp & (Cc - 1);
            int hh  = tmp >> 5;
            int gh  = h0 - 1 + hh;
            int gw  = w0 - 1 + ww;
            float v = 0.0f;
            if ((unsigned)gh < (unsigned)Hh && (unsigned)gw < (unsigned)Ww)
                v = ypb[(size_t)c * HWp + (size_t)gh * Ww + gw];
            yt[((size_t)hh * LW + ww) * Cc + c] = (bf16)v;
        }
    }
    __syncthreads();

    const int gh = wv;            // pixel row of this wave within tile
    const int n  = lane & 15;     // pixel within 16-wide group (N index)
    const int kg = lane >> 4;     // lane half (C/D: M-offset 8; B: K-offset 16)
    const int c0 = kg * 16;       // first input channel this lane supplies to B

    const float* sb = s + (size_t)b * HWp;

    for (int half = 0; half < 2; ++half) {
        const int wbase = half * 16;
        v8f acc0 = {};
        v8f acc1 = {};

        if (!first) {
#pragma unroll
            for (int tap = 0; tap < 9; ++tap) {
                const int dy = tap / 3 - 1;
                const int dx = tap % 3 - 1;
                // A fragments: pre-swizzled weights, coalesced 32B per lane.
                v16bf a0 = *(const v16bf*)(wA + ((size_t)(tap * 2 + 0) * 32 + lane) * 16);
                v16bf a1 = *(const v16bf*)(wA + ((size_t)(tap * 2 + 1) * 32 + lane) * 16);
                // B fragment (32x16 KxN): lane supplies 16 consecutive channels
                // at pixel (gh+dy, wbase+n+dx); lanes 0-15 K=0..15, 16-31 K=16..31.
                const int hh = gh + 1 + dy;
                const int ww = wbase + n + 1 + dx;
                v16bf bf = *(const v16bf*)&yt[((size_t)hh * LW + ww) * Cc + c0];
                acc0 = __builtin_amdgcn_wmma_f32_16x16x32_bf16(
                    false, a0, false, bf, (short)0, acc0, false, false);
                acc1 = __builtin_amdgcn_wmma_f32_16x16x32_bf16(
                    false, a1, false, bf, (short)0, acc1, false, false);
            }
        }

        // --- fused recurrence epilogue on the accumulator layout ---
        const int h = h0 + gh;
        const int w = w0 + wbase + n;

        // l-path stencil on channel-sum (same value for every channel).
        float lc = 0.0f;
        if (!first) {
            auto S = [&](int hh2, int ww2) -> float {
                if ((unsigned)hh2 < (unsigned)Hh && (unsigned)ww2 < (unsigned)Ww)
                    return sb[(size_t)hh2 * Ww + ww2];
                return 0.0f;
            };
            lc = 0.5f * (S(h - 1, w - 1) + S(h - 1, w + 1) +
                         S(h + 1, w - 1) + S(h + 1, w + 1)) +
                 (S(h - 1, w) + S(h, w - 1) + S(h, w + 1) + S(h + 1, w));
        }

#pragma unroll
        for (int ai = 0; ai < 2; ++ai) {
            v8f a = ai ? acc1 : acc0;
#pragma unroll
            for (int r = 0; r < 8; ++r) {
                const int co = r + 8 * kg + 16 * ai;          // output channel
                const size_t si = ((size_t)(b * Cc + co)) * HWp + (size_t)h * Ww + w;
                const size_t pi = (size_t)co * HWp + (size_t)h * Ww + w;

                float fv = DF * fbuf[si] + a[r] + bias[co] + xt[pi];
                float lv = DL * lbuf[si] + lc;
                float u  = fv + 0.5f * fv * lv;
                float yp = first ? 0.0f : ypb[pi];
                float ev = DE * ebuf[si] + VE * yp;
                float yv = 1.0f / (1.0f + __expf(ev - u));    // sigmoid(u - e)

                fbuf[si] = fv;
                lbuf[si] = lv;
                ebuf[si] = ev;
                yob[pi]  = yv;
            }
        }
    }
}

// ---------------------------------------------------------------------------
// Host launcher: init states, swizzle weights, then 10 recurrent steps.
// All launches on `stream`; no sync/alloc -> graph-capture safe.
// ---------------------------------------------------------------------------
extern "C" void kernel_launch(void* const* d_in, const int* in_sizes, int n_in,
                              void* d_out, int out_size, void* d_ws, size_t ws_size,
                              hipStream_t stream) {
    const float* x      = (const float*)d_in[0];  // [B,T,C,H,W]
    const float* conv_w = (const float*)d_in[1];  // [C,C,3,3]
    const float* conv_b = (const float*)d_in[2];  // [C]
    float*       out    = (float*)d_out;          // [B,T,C,H,W]

    const size_t N1 = (size_t)Bb * Cc * HWp;      // per-state element count
    float* f = (float*)d_ws;
    float* l = f + N1;
    float* e = l + N1;
    float* s = e + N1;                            // [B,H,W] channel sums
    bf16*  wA = (bf16*)(s + (size_t)Bb * HWp);    // 9*2*32*16 bf16 (32B aligned)

    init_state_kernel<<<(unsigned)((N1 + 255) / 256), 256, 0, stream>>>(f, l, e);
    prep_weights_kernel<<<(9 * 2 * 32 * 16 + 255) / 256, 256, 0, stream>>>(conv_w, wA);

    dim3 grid(Ww / TW, Hh / TH, Bb);              // (4, 16, 8)
    for (int t = 0; t < Tt; ++t) {
        if (t > 0) {
            chan_sum_kernel<<<(unsigned)(((size_t)Bb * HWp + 255) / 256), 256, 0, stream>>>(
                out, s, t - 1);
        }
        step_kernel<<<grid, 256, 0, stream>>>(x, conv_b, wA, s, out, f, l, e,
                                              t, (t == 0) ? 1 : 0);
    }
}
